// EvMLP_33964601377094
// MI455X (gfx1250) — matrix operand order repaired and checked by reading, not compiled, not run on hardware
//
#include <hip/hip_runtime.h>
#include <stdint.h>

typedef __attribute__((ext_vector_type(16))) __bf16 v16bf;
typedef __attribute__((ext_vector_type(8)))  float  v8f;

#define NROWS   100000
#define DIM     592
#define NINV    128
#define K1      240
#define WPB     8                      // waves per block
#define TILES   ((NROWS + 15) / 16)    // 6250

// LDS layout (dwords)
#define W1F_DW  16384                  // [8 nt][8 ks][32 lane][8 dw]
#define W2F_DW  8192                   // [8 nt][4 ks][32 lane][8 dw]
#define W3F_DW  8192
#define WGT_DW  (W1F_DW + W2F_DW + W3F_DW)
#define XAB_DW  2048                   // per-wave A staging: bf16 [16][256]
#define HB_DW   2048                   // per-wave f32 [16][128]
#define PW_DW   (XAB_DW + HB_DW)
#define TOTAL_DW (WGT_DW + WPB * PW_DW)   // 65536 dwords = 256 KB

__device__ __forceinline__ unsigned short f2bf(float f) {
  unsigned u = __float_as_uint(f);
  u += 0x7FFFu + ((u >> 16) & 1u);     // round-to-nearest-even
  return (unsigned short)(u >> 16);
}
__device__ __forceinline__ uint32_t packbf(float lo, float hi) {
  return (uint32_t)f2bf(lo) | ((uint32_t)f2bf(hi) << 16);
}

union FragU { uint32_t u[8]; uint4 q[2]; v16bf v; };

// A (16x32 bf16) fragment for lane (m, half) at k-step ks, from row-major bf16 [16][256]
__device__ __forceinline__ v16bf load_afrag(const uint32_t* xab, int m, int half, int ks) {
  FragU f;
  const uint4* p = (const uint4*)(xab + m * 128 + ks * 16 + half * 8);
  f.q[0] = p[0];   // dwords +0..3   (K = ks*32 + half*16 + 0..7)
  f.q[1] = p[1];   // dwords +4..7   (K = ks*32 + half*16 + 8..15)
  return f.v;
}
// B (32x16 bf16) fragment: lane's 8 dwords stored contiguously
__device__ __forceinline__ v16bf load_bfrag(const uint32_t* wf, int idx) {
  FragU f;
  const uint4* p = (const uint4*)(wf + idx * 8);
  f.q[0] = p[0];
  f.q[1] = p[1];
  return f.v;
}

__device__ __forceinline__ void layernorm_to_bf16(const float* hrow,         // hbuf + m*128
                                                  unsigned short* xrow,      // bf16 row, 256 cols
                                                  int half,
                                                  const float* __restrict__ g,
                                                  const float* __restrict__ be) {
  const float* hr = hrow + half * 64;
  float s = 0.f, sq = 0.f;
#pragma unroll 8
  for (int j = 0; j < 64; ++j) { float x = hr[j]; s += x; sq += x * x; }
  s  += __shfl_xor(s, 16);
  sq += __shfl_xor(sq, 16);
  float mu  = s  * (1.f / 128.f);
  float var = sq * (1.f / 128.f) - mu * mu;
  float rs  = rsqrtf(var + 1e-5f);
#pragma unroll 8
  for (int j = 0; j < 64; ++j) {
    int c = half * 64 + j;
    float y = (hr[j] - mu) * rs * g[c] + be[c];
    xrow[c] = f2bf(y);
  }
}

#define LDS_FENCE() asm volatile("s_wait_dscnt 0x0" ::: "memory")

#define GEMM_STEPS(WF, KS)                                                        \
  for (int ks = 0; ks < (KS); ++ks) {                                             \
    v16bf a = load_afrag(xab, m, half, ks);                                       \
    _Pragma("unroll")                                                             \
    for (int nt = 0; nt < 8; ++nt) {                                              \
      v16bf b = load_bfrag((WF), (nt * (KS) + ks) * 32 + lane);                   \
      acc[nt] = __builtin_amdgcn_wmma_f32_16x16x32_bf16(                          \
          false, a, false, b, (short)0, acc[nt], false, false);                   \
    }                                                                             \
  }

__global__ __launch_bounds__(256, 1) void evmlp_kernel(
    const float* __restrict__ ten,
    const float* __restrict__ w1, const float* __restrict__ g1, const float* __restrict__ be1,
    const float* __restrict__ w2, const float* __restrict__ b2,
    const float* __restrict__ g2, const float* __restrict__ be2,
    const float* __restrict__ w3, const float* __restrict__ b3,
    float* __restrict__ out) {
  extern __shared__ uint32_t lds[];
  const int tid = threadIdx.x;

  // ---- cooperative weight -> bf16 B-fragment conversion ----
  // w1: K padded 240->256, [nt][ks][lane][v]; element (lane,v): n = nt*16+(lane&15),
  // k = ks*32 + (lane>>4)*16 + 2v (+1 for hi half)
  for (int i = tid; i < W1F_DW; i += 256) {
    int v = i & 7, lane = (i >> 3) & 31, ks = (i >> 8) & 7, nt = i >> 11;
    int n = nt * 16 + (lane & 15);
    int k = ks * 32 + (lane >> 4) * 16 + 2 * v;
    float lo = (k     < K1) ? w1[(size_t)k       * NINV + n] : 0.f;
    float hi = (k + 1 < K1) ? w1[(size_t)(k + 1) * NINV + n] : 0.f;
    lds[i] = packbf(lo, hi);
  }
  for (int i = tid; i < W2F_DW; i += 256) {
    int v = i & 7, lane = (i >> 3) & 31, ks = (i >> 8) & 3, nt = i >> 10;
    int n = nt * 16 + (lane & 15);
    int k = ks * 32 + (lane >> 4) * 16 + 2 * v;
    lds[W1F_DW + i] = packbf(w2[(size_t)k * NINV + n], w2[(size_t)(k + 1) * NINV + n]);
  }
  for (int i = tid; i < W3F_DW; i += 256) {
    int v = i & 7, lane = (i >> 3) & 31, ks = (i >> 8) & 3, nt = i >> 10;
    int n = nt * 16 + (lane & 15);
    int k = ks * 32 + (lane >> 4) * 16 + 2 * v;
    lds[W1F_DW + W2F_DW + i] = packbf(w3[(size_t)k * NINV + n], w3[(size_t)(k + 1) * NINV + n]);
  }
  __syncthreads();

  const int wave = tid >> 5;
  const int lane = tid & 31;
  const int half = lane >> 4;   // 2 lanes per row
  const int m    = lane & 15;   // row within tile
  const int tile = blockIdx.x * WPB + wave;
  if (tile >= TILES) return;

  uint32_t* xab = lds + WGT_DW + wave * PW_DW;
  float*    hbuf = (float*)(xab + XAB_DW);
  unsigned short* xab16 = (unsigned short*)xab;

  const int row = tile * 16 + m;
  const float* rp = ten + (size_t)row * DIM;
  float*       op = out + (size_t)row * DIM;

  // ---- phase 1: invariants + channel norms; build x1 (bf16, A-layout rows), emit x2 ----
  if (half == 0) {
    // invariant cols 0..127 -> xab dwords [m][0..63]
    const float4* r4 = (const float4*)rp;
#pragma unroll 8
    for (int j = 0; j < 32; ++j) {
      float4 f = r4[j];
      xab[m * 128 + 2 * j]     = packbf(f.x, f.y);
      xab[m * 128 + 2 * j + 1] = packbf(f.z, f.w);
    }
    // l=1 channels 0..55 (cols 128..295)
    for (int c = 0; c < 56; ++c) {
      int col = 128 + 3 * c;
      float a = rp[col], b = rp[col + 1], d = rp[col + 2];
      float dv = sqrtf(a * a + b * b + d * d + 1.f);
      float rv = 1.f / dv;
      op[col] = a * rv; op[col + 1] = b * rv; op[col + 2] = d * rv;
      xab16[m * 256 + 128 + c] = f2bf(dv - 1.f);
    }
    for (int c = 240; c < 248; ++c) xab16[m * 256 + c] = 0;  // K padding
  } else {
    // l=1 channels 56..63 (cols 296..319)
    for (int c = 56; c < 64; ++c) {
      int col = 128 + 3 * c;
      float a = rp[col], b = rp[col + 1], d = rp[col + 2];
      float dv = sqrtf(a * a + b * b + d * d + 1.f);
      float rv = 1.f / dv;
      op[col] = a * rv; op[col + 1] = b * rv; op[col + 2] = d * rv;
      xab16[m * 256 + 128 + c] = f2bf(dv - 1.f);
    }
    // l=2: 32 channels x 5 (cols 320..479), x11 ids 64..95
    for (int c = 0; c < 32; ++c) {
      int col = 320 + 5 * c;
      float x[5], ss = 0.f;
#pragma unroll
      for (int j = 0; j < 5; ++j) { x[j] = rp[col + j]; ss += x[j] * x[j]; }
      float dv = sqrtf(ss + 1.f), rv = 1.f / dv;
#pragma unroll
      for (int j = 0; j < 5; ++j) op[col + j] = x[j] * rv;
      xab16[m * 256 + 128 + 64 + c] = f2bf(dv - 1.f);
    }
    // l=3: 16 channels x 7 (cols 480..591), x11 ids 96..111
    for (int c = 0; c < 16; ++c) {
      int col = 480 + 7 * c;
      float x[7], ss = 0.f;
#pragma unroll
      for (int j = 0; j < 7; ++j) { x[j] = rp[col + j]; ss += x[j] * x[j]; }
      float dv = sqrtf(ss + 1.f), rv = 1.f / dv;
#pragma unroll
      for (int j = 0; j < 7; ++j) op[col + j] = x[j] * rv;
      xab16[m * 256 + 128 + 96 + c] = f2bf(dv - 1.f);
    }
    for (int c = 248; c < 256; ++c) xab16[m * 256 + c] = 0;  // K padding
  }
  LDS_FENCE();

  v8f acc[8];

  // ---- GEMM1: h = x1 @ w1  (K=256 padded, 64 WMMA) ----
#pragma unroll
  for (int nt = 0; nt < 8; ++nt) {
    v8f z; for (int r = 0; r < 8; ++r) z[r] = 0.f; acc[nt] = z;
  }
  GEMM_STEPS(lds, 8);
#pragma unroll
  for (int nt = 0; nt < 8; ++nt)
#pragma unroll
    for (int r = 0; r < 8; ++r)
      hbuf[(r + 8 * half) * 128 + nt * 16 + m] = acc[nt][r];
  LDS_FENCE();
  layernorm_to_bf16(hbuf + m * 128, xab16 + m * 256, half, g1, be1);
  LDS_FENCE();

  // ---- GEMM2: h = LN1 @ w2 + b2, then SiLU (32 WMMA) ----
#pragma unroll
  for (int nt = 0; nt < 8; ++nt) {
    float bv = b2[nt * 16 + m];
    v8f z; for (int r = 0; r < 8; ++r) z[r] = bv; acc[nt] = z;
  }
  GEMM_STEPS(lds + W1F_DW, 4);
#pragma unroll
  for (int nt = 0; nt < 8; ++nt)
#pragma unroll
    for (int r = 0; r < 8; ++r) {
      float f = acc[nt][r];
      hbuf[(r + 8 * half) * 128 + nt * 16 + m] = f / (1.f + __expf(-f));  // silu
    }
  LDS_FENCE();
  layernorm_to_bf16(hbuf + m * 128, xab16 + m * 256, half, g2, be2);
  LDS_FENCE();

  // ---- GEMM3: h = LN2 @ w3 + b3 (32 WMMA), store to out cols 0..127 ----
#pragma unroll
  for (int nt = 0; nt < 8; ++nt) {
    float bv = b3[nt * 16 + m];
    v8f z; for (int r = 0; r < 8; ++r) z[r] = bv; acc[nt] = z;
  }
  GEMM_STEPS(lds + W1F_DW + W2F_DW, 4);
#pragma unroll
  for (int nt = 0; nt < 8; ++nt)
#pragma unroll
    for (int r = 0; r < 8; ++r)
      out[(size_t)(tile * 16 + r + 8 * half) * DIM + nt * 16 + m] = acc[nt][r];
}

extern "C" void kernel_launch(void* const* d_in, const int* in_sizes, int n_in,
                              void* d_out, int out_size, void* d_ws, size_t ws_size,
                              hipStream_t stream) {
  const float* ten = (const float*)d_in[0];
  // d_in[1] = rep_layout (int64) -- structure is hardcoded, unused
  const float* w1  = (const float*)d_in[2];
  const float* g1  = (const float*)d_in[3];
  const float* be1 = (const float*)d_in[4];
  const float* w2  = (const float*)d_in[5];
  const float* b2  = (const float*)d_in[6];
  const float* g2  = (const float*)d_in[7];
  const float* be2 = (const float*)d_in[8];
  const float* w3  = (const float*)d_in[9];
  const float* b3  = (const float*)d_in[10];

  int blocks = (TILES + WPB - 1) / WPB;   // 782
  hipLaunchKernelGGL(evmlp_kernel, dim3(blocks), dim3(256),
                     TOTAL_DW * sizeof(uint32_t), stream,
                     ten, w1, g1, be1, w2, b2, g2, be2, w3, b3, (float*)d_out);
}
